// TransformerLayerShardV2_65816078844539
// MI455X (gfx1250) — matrix operand order repaired and checked by reading, not compiled, not run on hardware
//
#include <hip/hip_runtime.h>
#include <hip/hip_bf16.h>

// ---------------------------------------------------------------------------
// TransformerLayerShardV2 for MI455X (gfx1250): bf16 WMMA everywhere,
// async global->LDS double-buffered GEMM pipeline.
//   SEQ=2048, D=4096, N_HEADS=16, DH=256, ROT=64
// ---------------------------------------------------------------------------

#define SEQ     2048
#define DMODEL  4096
#define NHEADS  16
#define DH      256
#define ROT     64
#define PROJ_N  (7 * DMODEL)   // 28672
#define CAT_N   (5 * DMODEL)   // 20480
#define FF_N    (4 * DMODEL)   // 16384

typedef __attribute__((ext_vector_type(16))) __bf16 bf16x16;
typedef __attribute__((ext_vector_type(8)))  float  f32x8;

union Frag16 { bf16x16 v; uint4 q[2]; };

// --- CDNA5 async global->LDS path (guarded; falls back to sync copies) -----
#if defined(__gfx1250__) && __has_builtin(__builtin_amdgcn_global_load_async_to_lds_b128)
#define HAVE_ASYNC_LDS 1
#else
#define HAVE_ASYNC_LDS 0
#endif

#if HAVE_ASYNC_LDS
// Builtin signature (from clang diagnostic):
//   void (v4i __attribute__((addrspace(1)))* src, v4i addrspace(3)* dst,
//         imm int offset, imm int cpol)
// generic->AS1 / AS3 casts via uintptr_t: low 32 bits of a generic LDS pointer
// are the LDS byte offset.
typedef int cdna5_v4i __attribute__((vector_size(4 * sizeof(int))));
#define ASYNC_CP16(g, l)                                                       \
    __builtin_amdgcn_global_load_async_to_lds_b128(                            \
        (__attribute__((address_space(1))) cdna5_v4i*)(uintptr_t)(g),          \
        (__attribute__((address_space(3))) cdna5_v4i*)(uint32_t)(uintptr_t)(l),\
        0, 0)
#if __has_builtin(__builtin_amdgcn_s_wait_asynccnt)
#define WAIT_ASYNC(n) __builtin_amdgcn_s_wait_asynccnt(n)
#else
#define WAIT_ASYNC(n) asm volatile("s_wait_asynccnt %0" ::"n"(n) : "memory")
#endif
#else
#define WAIT_ASYNC(n) ((void)0)
#endif

// ---------------------------------------------------------------------------
// fp32 [R][C] -> bf16 transposed [C][R]   (tiled 32x32 through LDS)
// ---------------------------------------------------------------------------
__global__ __launch_bounds__(256)
void transpose_cast(const float* __restrict__ W, __bf16* __restrict__ Wt,
                    int R, int C) {
    __shared__ float tile[32][33];
    const int bc = blockIdx.x * 32;
    const int br = blockIdx.y * 32;
    const int tx = threadIdx.x & 31;
    const int ty = threadIdx.x >> 5;        // 0..7
#pragma unroll
    for (int p = 0; p < 4; ++p) {
        int r = ty + p * 8;
        tile[r][tx] = W[(size_t)(br + r) * C + (bc + tx)];
    }
    __syncthreads();
#pragma unroll
    for (int p = 0; p < 4; ++p) {
        int c = ty + p * 8;
        Wt[(size_t)(bc + c) * R + (br + tx)] = (__bf16)tile[tx][c];
    }
}

// ---------------------------------------------------------------------------
// LayerNorm row kernel: fp32 x -> bf16 xn
// ---------------------------------------------------------------------------
__global__ __launch_bounds__(256)
void layernorm_cast(const float* __restrict__ x,
                    const float* __restrict__ sc,
                    const float* __restrict__ of,
                    __bf16* __restrict__ xn) {
    __shared__ float red[256];
    const int row = blockIdx.x;
    const int tid = threadIdx.x;
    const float* xr = x + (size_t)row * DMODEL;

    float s = 0.f, ss = 0.f;
    for (int i = tid; i < DMODEL; i += 256) {
        float v = xr[i];
        s += v; ss += v * v;
    }
    red[tid] = s; __syncthreads();
    for (int o = 128; o > 0; o >>= 1) { if (tid < o) red[tid] += red[tid + o]; __syncthreads(); }
    float mean = red[0] * (1.f / DMODEL);
    __syncthreads();
    red[tid] = ss; __syncthreads();
    for (int o = 128; o > 0; o >>= 1) { if (tid < o) red[tid] += red[tid + o]; __syncthreads(); }
    float var = red[0] * (1.f / DMODEL) - mean * mean;
    float rs  = rsqrtf(var + 1e-5f);

    __bf16* orow = xn + (size_t)row * DMODEL;
    for (int i = tid; i < DMODEL; i += 256)
        orow[i] = (__bf16)(((xr[i] - mean) * rs) * sc[i] + of[i]);
}

// ---------------------------------------------------------------------------
// bf16 GEMM: C[M][N] = A[M][K] * Bt[N][K]^T, f32 accumulate via WMMA.
// Block tile 128x128x32, 256 threads = 8 waves (2x4), wave tile 64x32 (4x2).
// Double-buffered LDS, async global->LDS copies, prefetch distance 2.
//  EPI 0: fp32 C = acc + bias[n]
//  EPI 1: fp32 C = acc*scale + bias2d[m][n], masked -1e10 where n>m (causal)
//  EPI 2: bf16 C = acc
// ---------------------------------------------------------------------------
template <int EPI>
__global__ __launch_bounds__(256)
void gemm_bf16(const __bf16* __restrict__ A, const __bf16* __restrict__ Bt,
               void* __restrict__ Cv,
               const float* __restrict__ bias, const float* __restrict__ bias2d,
               int K, int lda, int ldb, int ldc, int ld2,
               long strideA, long strideB, long strideC, float scale) {
    __shared__ __attribute__((aligned(16))) __bf16 As[2][128 * 32];
    __shared__ __attribute__((aligned(16))) __bf16 Bs[2][128 * 32];

    const int tid  = threadIdx.x;
    const int lane = tid & 31;
    const int wave = tid >> 5;
    const int wm   = wave & 1;          // 2 waves along M
    const int wn   = wave >> 1;         // 4 waves along N
    const int bm   = blockIdx.y * 128;
    const int bn   = blockIdx.x * 128;
    const int bz   = blockIdx.z;

    A  += (size_t)bz * strideA;
    Bt += (size_t)bz * strideB;

    const int hf    = lane >> 4;        // 0/1 half-wave
    const int mlane = lane & 15;
    const int crow  = tid >> 1;         // 0..127
    const int ccol  = (tid & 1) * 16;   // 0 or 16 (x bf16)

    const __bf16* gA = A  + (size_t)(bm + crow) * lda + ccol;
    const __bf16* gB = Bt + (size_t)(bn + crow) * ldb + ccol;

    auto issue_copy = [&](int buf, int k0) {
        const __bf16* ga = gA + k0;
        const __bf16* gb = gB + k0;
        __bf16* la = &As[buf][crow * 32 + ccol];
        __bf16* lb = &Bs[buf][crow * 32 + ccol];
#if HAVE_ASYNC_LDS
        ASYNC_CP16(ga,     la);
        ASYNC_CP16(ga + 8, la + 8);
        ASYNC_CP16(gb,     lb);
        ASYNC_CP16(gb + 8, lb + 8);
#else
        const uint4* g0 = (const uint4*)ga;
        const uint4* g1 = (const uint4*)gb;
        uint4 a0 = g0[0], a1 = g0[1], b0 = g1[0], b1 = g1[1];
        uint4* l0 = (uint4*)la; l0[0] = a0; l0[1] = a1;
        uint4* l1 = (uint4*)lb; l1[0] = b0; l1[1] = b1;
#endif
    };

    f32x8 acc[4][2] = {};
    const int NT = K / 32;              // >= 8 for every GEMM in this layer

    issue_copy(0, 0);
    issue_copy(1, 32);

    for (int t = 0; t < NT; ++t) {
        const int buf = t & 1;
        if (t + 1 < NT) { WAIT_ASYNC(4); }   // tile t landed; t+1 may be in flight
        else            { WAIT_ASYNC(0); }
        __syncthreads();

        Frag16 af[4], bfm[2];
#pragma unroll
        for (int i = 0; i < 4; ++i) {   // ISA 7.12.2 16-bit A layout (16x32)
            const __bf16* p = &As[buf][(wm * 64 + i * 16 + mlane) * 32];
            af[i].q[0] = *(const uint4*)(p + hf * 8);        // K = hf*8 .. +7
            af[i].q[1] = *(const uint4*)(p + 16 + hf * 8);   // K = 16+hf*8 ..
        }
#pragma unroll
        for (int j = 0; j < 2; ++j) {   // B: lane holds col N, same K layout as A
            const __bf16* p = &Bs[buf][(wn * 32 + j * 16 + mlane) * 32];
            bfm[j].q[0] = *(const uint4*)(p + hf * 8);
            bfm[j].q[1] = *(const uint4*)(p + 16 + hf * 8);
        }
        __syncthreads();                // all reads of buf done -> safe to refill

        if (t + 2 < NT) issue_copy(buf, (t + 2) * 32);

#pragma unroll
        for (int i = 0; i < 4; ++i)
#pragma unroll
            for (int j = 0; j < 2; ++j)
                acc[i][j] = __builtin_amdgcn_wmma_f32_16x16x32_bf16(
                    false, af[i].v, false, bfm[j].v,
                    (short)0, acc[i][j], false, false);
    }

#pragma unroll
    for (int i = 0; i < 4; ++i)
#pragma unroll
        for (int j = 0; j < 2; ++j) {
            const int gm0 = bm + wm * 64 + i * 16 + hf * 8;  // C: vgpr r -> M=r+8*hf
            const int gn  = bn + wn * 32 + j * 16 + mlane;
#pragma unroll
            for (int r = 0; r < 8; ++r) {
                const int gm = gm0 + r;
                float v = acc[i][j][r];
                if constexpr (EPI == 0) {
                    float* C = (float*)Cv + (size_t)bz * strideC;
                    if (bias) v += bias[gn];
                    C[(size_t)gm * ldc + gn] = v;
                } else if constexpr (EPI == 1) {
                    float* C = (float*)Cv + (size_t)bz * strideC;
                    v = v * scale + bias2d[(size_t)gm * ld2 + gn];
                    if (gn > gm) v = -1e10f;                 // causal mask
                    C[(size_t)gm * ldc + gn] = v;
                } else {
                    __bf16* C = (__bf16*)Cv + (size_t)bz * strideC;
                    C[(size_t)gm * ldc + gn] = (__bf16)v;
                }
            }
        }
}

// ---------------------------------------------------------------------------
// RoPE + pack: proj fp32 -> Qb[h][s][256], Kb[h][s][256], Vt[h][256][s] (bf16)
// ---------------------------------------------------------------------------
__global__ __launch_bounds__(256)
void rope_pack(const float* __restrict__ proj,
               __bf16* __restrict__ Qb, __bf16* __restrict__ Kb,
               __bf16* __restrict__ Vt) {
    const int s = blockIdx.x;
    const int h = blockIdx.y;
    const int d = threadIdx.x;          // 0..255
    const float* row = proj + (size_t)s * PROJ_N;

    float q = row[h * DH + d];
    float v = row[DMODEL + h * DH + d];
    float k = row[2 * DMODEL + h * DH + d];

    if (d < ROT) {
        const int i = d >> 1;
        float inv = expf(-((float)(2 * i) / (float)ROT) * 9.210340371976184f); // ln 1e4
        float ang = (float)s * inv;
        float sn = sinf(ang), cs = cosf(ang);
        float qp = row[h * DH + (d ^ 1)];
        float kp = row[2 * DMODEL + h * DH + (d ^ 1)];
        if (d & 1) { q = q * cs + qp * sn; k = k * cs + kp * sn; }
        else       { q = q * cs - qp * sn; k = k * cs - kp * sn; }
    }
    const size_t qi = ((size_t)h * SEQ + s) * DH + d;
    Qb[qi] = (__bf16)q;
    Kb[qi] = (__bf16)k;
    Vt[((size_t)h * DH + d) * SEQ + s] = (__bf16)v;
}

// ---------------------------------------------------------------------------
// Row softmax over 2048 cols: fp32 scores -> bf16 probs
// ---------------------------------------------------------------------------
__global__ __launch_bounds__(256)
void softmax_rows(const float* __restrict__ S, __bf16* __restrict__ P) {
    __shared__ float red[256];
    const size_t row = blockIdx.x;      // h*SEQ + s
    const int tid = threadIdx.x;
    const float* sr = S + row * SEQ;
    __bf16* pr = P + row * SEQ;

    float m = -3.4e38f;
    float loc[8];
#pragma unroll
    for (int p = 0; p < 8; ++p) {
        loc[p] = sr[tid + p * 256];
        m = fmaxf(m, loc[p]);
    }
    red[tid] = m; __syncthreads();
    for (int o = 128; o > 0; o >>= 1) { if (tid < o) red[tid] = fmaxf(red[tid], red[tid + o]); __syncthreads(); }
    m = red[0]; __syncthreads();

    float s = 0.f;
#pragma unroll
    for (int p = 0; p < 8; ++p) { loc[p] = expf(loc[p] - m); s += loc[p]; }
    red[tid] = s; __syncthreads();
    for (int o = 128; o > 0; o >>= 1) { if (tid < o) red[tid] += red[tid + o]; __syncthreads(); }
    const float inv = 1.f / red[0];
#pragma unroll
    for (int p = 0; p < 8; ++p)
        pr[tid + p * 256] = (__bf16)(loc[p] * inv);
}

// ---------------------------------------------------------------------------
// GELU (tanh approx, jax.nn.gelu default) on ff -> bf16 into concat cols 4096+
// ---------------------------------------------------------------------------
__global__ __launch_bounds__(256)
void gelu_cast(const float* __restrict__ proj, __bf16* __restrict__ cat) {
    const size_t i = (size_t)blockIdx.x * 256 + threadIdx.x;  // over SEQ*FF_N
    const int s = (int)(i / FF_N);
    const int c = (int)(i % FF_N);
    float v = proj[(size_t)s * PROJ_N + 3 * DMODEL + c];
    float g = 0.5f * v * (1.f + tanhf(0.7978845608028654f * (v + 0.044715f * v * v * v)));
    cat[(size_t)s * CAT_N + DMODEL + c] = (__bf16)g;
}

// ---------------------------------------------------------------------------
extern "C" void kernel_launch(void* const* d_in, const int* in_sizes, int n_in,
                              void* d_out, int out_size, void* d_ws, size_t ws_size,
                              hipStream_t stream) {
    const float* x        = (const float*)d_in[0];
    const float* attn_b   = (const float*)d_in[1];
    const float* ln_sc    = (const float*)d_in[2];
    const float* ln_of    = (const float*)d_in[3];
    const float* w_in     = (const float*)d_in[4];
    const float* b_in     = (const float*)d_in[5];
    const float* w_out    = (const float*)d_in[6];
    const float* b_out    = (const float*)d_in[7];

    char* ws = (char*)d_ws;
    __bf16* xn      = (__bf16*)ws;  ws += (size_t)SEQ * DMODEL * 2;
    __bf16* w_in_t  = (__bf16*)ws;  ws += (size_t)PROJ_N * DMODEL * 2;
    __bf16* w_out_t = (__bf16*)ws;  ws += (size_t)DMODEL * CAT_N * 2;
    float*  proj    = (float*)ws;   ws += (size_t)SEQ * PROJ_N * 4;
    __bf16* Qb      = (__bf16*)ws;  ws += (size_t)NHEADS * SEQ * DH * 2;
    __bf16* Kb      = (__bf16*)ws;  ws += (size_t)NHEADS * SEQ * DH * 2;
    __bf16* Vt      = (__bf16*)ws;  ws += (size_t)NHEADS * SEQ * DH * 2;
    float*  scores  = (float*)ws;   ws += (size_t)NHEADS * SEQ * SEQ * 4;
    __bf16* probs   = (__bf16*)ws;  ws += (size_t)NHEADS * SEQ * SEQ * 2;
    __bf16* cat     = (__bf16*)ws;  ws += (size_t)SEQ * CAT_N * 2;

    // weights fp32 -> bf16 transposed ([N][K] so GEMM B-tiles are K-contiguous)
    transpose_cast<<<dim3(PROJ_N / 32, DMODEL / 32), 256, 0, stream>>>(w_in, w_in_t, DMODEL, PROJ_N);
    transpose_cast<<<dim3(DMODEL / 32, CAT_N / 32), 256, 0, stream>>>(w_out, w_out_t, CAT_N, DMODEL);

    layernorm_cast<<<SEQ, 256, 0, stream>>>(x, ln_sc, ln_of, xn);

    // proj = xn @ w_in + b_in   (2048 x 28672, K=4096)
    gemm_bf16<0><<<dim3(PROJ_N / 128, SEQ / 128, 1), 256, 0, stream>>>(
        xn, w_in_t, proj, b_in, nullptr,
        DMODEL, DMODEL, DMODEL, PROJ_N, 0, 0, 0, 0, 1.f);

    rope_pack<<<dim3(SEQ, NHEADS), 256, 0, stream>>>(proj, Qb, Kb, Vt);

    // scores[h] = Q K^T / 16 + attn_bias, causal-masked  (16 x 2048 x 2048, K=256)
    gemm_bf16<1><<<dim3(SEQ / 128, SEQ / 128, NHEADS), 256, 0, stream>>>(
        Qb, Kb, scores, nullptr, attn_b,
        DH, DH, DH, SEQ, SEQ,
        (long)SEQ * DH, (long)SEQ * DH, (long)SEQ * SEQ, 0.0625f);

    softmax_rows<<<NHEADS * SEQ, 256, 0, stream>>>(scores, probs);

    // attn_vec[h] = P @ V -> bf16 into concat cols [h*256, h*256+256)
    gemm_bf16<2><<<dim3(DH / 128, SEQ / 128, NHEADS), 256, 0, stream>>>(
        probs, Vt, cat, nullptr, nullptr,
        SEQ, SEQ, SEQ, CAT_N, 0,
        (long)SEQ * SEQ, (long)DH * SEQ, (long)DH, 1.f);

    gelu_cast<<<(SEQ * FF_N) / 256, 256, 0, stream>>>(proj, cat);

    // out = concat @ w_out + b_out  (2048 x 4096, K=20480)
    gemm_bf16<0><<<dim3(DMODEL / 128, SEQ / 128, 1), 256, 0, stream>>>(
        cat, w_out_t, (float*)d_out, b_out, nullptr,
        CAT_N, CAT_N, CAT_N, DMODEL, 0, 0, 0, 0, 1.f);
}
